// Zig_MambaBlock_86569360818674
// MI455X (gfx1250) — compile-verified, matrix operand order checked by reading, not run on hardware
//
#include <hip/hip_runtime.h>
#include <math.h>

// ---------------------------------------------------------------------------
// Problem constants (match reference)
// ---------------------------------------------------------------------------
#define B_ 16
#define S_ 128
#define D_ 1152
#define E_ 2304
#define R_ 72
#define N_ 16
#define M_ (B_ * S_)          // 2048 rows
#define THREED (3 * D_)       // 3456
#define TWOD (2 * D_)         // 2304

typedef __attribute__((ext_vector_type(16))) __bf16 v16bf;
typedef __attribute__((ext_vector_type(8)))  float  v8f;
typedef __attribute__((ext_vector_type(4)))  float  f32x4;
typedef __attribute__((ext_vector_type(4)))  int    v4i;

// Async global->LDS path (CDNA5). Guarded: falls back to sync staging if the
// toolchain doesn't declare the builtins.
#if defined(__has_builtin)
#if __has_builtin(__builtin_amdgcn_global_load_async_to_lds_b128) && \
    __has_builtin(__builtin_amdgcn_s_wait_asynccnt)
#define HAVE_ASYNC_LDS 1
#endif
#endif
#ifndef HAVE_ASYNC_LDS
#define HAVE_ASYNC_LDS 0
#endif

#if HAVE_ASYNC_LDS
// builtin expects: (v4i addrspace(1)*, v4i addrspace(3)*, imm offset, imm cpol)
typedef __attribute__((address_space(1))) v4i* gas1_v4i;
typedef __attribute__((address_space(3))) v4i* gas3_v4i;
#endif

// round-to-nearest-even f32 -> bf16
static __device__ __forceinline__ __bf16 f2bf(float f) {
    unsigned u = __builtin_bit_cast(unsigned, f);
    u += 0x7fffu + ((u >> 16) & 1u);
    unsigned short h = (unsigned short)(u >> 16);
    return __builtin_bit_cast(__bf16, h);
}

static __device__ __forceinline__ float siluf(float v) {
    return v / (1.f + __expf(-v));
}

// ---------------------------------------------------------------------------
// f32 -> bf16 conversion (weights)
// ---------------------------------------------------------------------------
__global__ void cvt_kernel(const float* __restrict__ in, __bf16* __restrict__ out, int n) {
    int i = blockIdx.x * blockDim.x + threadIdx.x;
    int stride = gridDim.x * blockDim.x;
    for (; i < n; i += stride) out[i] = f2bf(in[i]);
}

// ---------------------------------------------------------------------------
// mod = silu(c) @ W_ada.T + b_ada   (16 x 3456, K=2304) — tiny, VALU
// ---------------------------------------------------------------------------
__global__ void mod_kernel(const float* __restrict__ c, const float* __restrict__ W_ada,
                           const float* __restrict__ b_ada, float* __restrict__ modv) {
    int o = blockIdx.x * blockDim.x + threadIdx.x;
    if (o >= B_ * THREED) return;
    int bi = o / THREED, oo = o % THREED;
    const float* crow = c + (size_t)bi * TWOD;
    const float* wrow = W_ada + (size_t)oo * TWOD;
    float acc = 0.f;
    for (int j = 0; j < TWOD; j += 4) {
        float4 cv = *(const float4*)(crow + j);
        float4 wv = *(const float4*)(wrow + j);
        acc += siluf(cv.x) * wv.x + siluf(cv.y) * wv.y +
               siluf(cv.z) * wv.z + siluf(cv.w) * wv.w;
    }
    modv[o] = acc + b_ada[oo];
}

// ---------------------------------------------------------------------------
// block reduction helper (256 threads)
// ---------------------------------------------------------------------------
static __device__ __forceinline__ float block_reduce(float v, float* sred) {
    sred[threadIdx.x] = v;
    __syncthreads();
    for (int off = 128; off > 0; off >>= 1) {
        if ((int)threadIdx.x < off) sred[threadIdx.x] += sred[threadIdx.x + off];
        __syncthreads();
    }
    float r = sred[0];
    __syncthreads();
    return r;
}

// ---------------------------------------------------------------------------
// Fused: LN1 -> modulate(shift,scale) -> skip(f32) -> LN2 -> xn2 (bf16)
// one block per row (2048 blocks)
// ---------------------------------------------------------------------------
__global__ __launch_bounds__(256) void ln_mod_kernel(
    const float* __restrict__ x, const float* __restrict__ modv,
    const float* __restrict__ g1, const float* __restrict__ be1,
    const float* __restrict__ g2, const float* __restrict__ be2,
    float* __restrict__ skip, __bf16* __restrict__ xn2b)
{
    __shared__ float srow[D_];
    __shared__ float sred[256];
    int row = blockIdx.x;
    int b = row / S_;
    const float* xr = x + (size_t)row * D_;
    const float* shiftp = modv + (size_t)b * THREED;
    const float* scalep = shiftp + D_;

    float s1 = 0.f, s2 = 0.f;
    for (int i = threadIdx.x; i < D_; i += 256) {
        float v = xr[i];
        srow[i] = v;
        s1 += v; s2 += v * v;
    }
    float tsum = block_reduce(s1, sred);
    float tsq  = block_reduce(s2, sred);
    float mean = tsum * (1.f / D_);
    float var  = tsq * (1.f / D_) - mean * mean;
    float rs   = rsqrtf(var + 1e-5f);

    float p1 = 0.f, p2 = 0.f;
    for (int i = threadIdx.x; i < D_; i += 256) {
        float v  = srow[i];
        float xn = (v - mean) * rs * g1[i] + be1[i];
        float xm = xn * (1.f + scalep[i]) + shiftp[i];
        skip[(size_t)row * D_ + i] = xm;
        srow[i] = xm;
        p1 += xm; p2 += xm * xm;
    }
    float t1 = block_reduce(p1, sred);
    float t2 = block_reduce(p2, sred);
    float m2  = t1 * (1.f / D_);
    float v2  = t2 * (1.f / D_) - m2 * m2;
    float rs2 = rsqrtf(v2 + 1e-5f);

    for (int i = threadIdx.x; i < D_; i += 256) {
        float xm = srow[i];
        float xn2 = (xm - m2) * rs2 * g2[i] + be2[i];
        xn2b[(size_t)row * D_ + i] = f2bf(xn2);
    }
}

// ---------------------------------------------------------------------------
// WMMA GEMM with LDS-staged, double-buffered tiles.
//   C[M x Nn] = A[M x K] * W[Nn x K]^T (+ epilogue)
// Block: 256 threads = 8 waves; block tile 128x128; wave tile 32x64.
// K-step 32. LDS row pitch padded to 40 bf16 (80 B) -> bank-conflict-free
// b128 fragment reads (row r starts at bank 20*r mod 64, distinct r=0..15).
// ---------------------------------------------------------------------------
#define LDSK 40

// Per ISA 16-bit A 16x32 layout: lanes 0-15 -> rows, K 0..7 / 16..23;
// lanes 16-31 -> same rows, K 8..15 / 24..31. B loads symmetrically.
static __device__ __forceinline__ v16bf load_frag_lds(const __bf16* buf, int row0, int lane) {
    int half = lane >> 4;
    int r    = lane & 15;
    const __bf16* p = buf + (row0 + r) * LDSK + half * 8;
    union { v16bf v; f32x4 q[2]; } f;
    f.q[0] = *(const f32x4*)p;
    f.q[1] = *(const f32x4*)(p + 16);
    return f.v;
}

// Stage one 128x32 tile pair (A and W) into LDS. Each of the 256 threads
// moves one 32-byte half-row of A and of W (2x b128 each).
static __device__ __forceinline__ void stage_tiles(
    const __bf16* __restrict__ gA, int ldA,
    const __bf16* __restrict__ gB, int ldB,
    __bf16* bufA, __bf16* bufB, int tid)
{
    int row = tid >> 1;
    int seg = (tid & 1) * 16;               // element offset of 32-byte chunk
    const __bf16* ga = gA + (size_t)row * ldA + seg;
    const __bf16* gb = gB + (size_t)row * ldB + seg;
    __bf16* la = bufA + row * LDSK + seg;
    __bf16* lb = bufB + row * LDSK + seg;
#if HAVE_ASYNC_LDS
    __builtin_amdgcn_global_load_async_to_lds_b128((gas1_v4i)ga,       (gas3_v4i)la,       0, 0);
    __builtin_amdgcn_global_load_async_to_lds_b128((gas1_v4i)(ga + 8), (gas3_v4i)(la + 8), 0, 0);
    __builtin_amdgcn_global_load_async_to_lds_b128((gas1_v4i)gb,       (gas3_v4i)lb,       0, 0);
    __builtin_amdgcn_global_load_async_to_lds_b128((gas1_v4i)(gb + 8), (gas3_v4i)(lb + 8), 0, 0);
#else
    f32x4 a0 = *(const f32x4*)ga;
    f32x4 a1 = *(const f32x4*)(ga + 8);
    f32x4 b0 = *(const f32x4*)gb;
    f32x4 b1 = *(const f32x4*)(gb + 8);
    *(f32x4*)la       = a0;
    *(f32x4*)(la + 8) = a1;
    *(f32x4*)lb       = b0;
    *(f32x4*)(lb + 8) = b1;
#endif
}

static __device__ __forceinline__ void tiles_ready_barrier() {
#if HAVE_ASYNC_LDS
    __builtin_amdgcn_s_wait_asynccnt(0);
#endif
    __syncthreads();
}

// MODE 0: out_f32  = acc + bias
// MODE 1: out_bf16 = bf16(acc + bias)
// MODE 3: out_f32  = x + gate[b]*(acc + bias + skip)
template <int MODE>
__global__ __launch_bounds__(256) void gemm_bf16_kernel(
    const __bf16* __restrict__ A, const __bf16* __restrict__ W,
    int M, int Nn, int K,
    const float* __restrict__ bias,
    float* __restrict__ out_f32, __bf16* __restrict__ out_bf16,
    const float* __restrict__ x_res, const float* __restrict__ modv,
    const float* __restrict__ skip)
{
    __shared__ __align__(16) __bf16 ldsA[2][128 * LDSK];
    __shared__ __align__(16) __bf16 ldsB[2][128 * LDSK];

    int tiles_n = Nn >> 7;
    int tm = blockIdx.x / tiles_n;
    int tn = blockIdx.x % tiles_n;
    int tid  = threadIdx.x;
    int w    = tid >> 5;
    int lane = tid & 31;
    int wm = w & 3;        // 4 M-slots of 32
    int wn = w >> 2;       // 2 N-slots of 64
    int lm_base = wm * 32; // local (LDS) row bases
    int ln_base = wn * 64;

    const __bf16* gA = A + (size_t)(tm * 128) * K;
    const __bf16* gB = W + (size_t)(tn * 128) * K;

    v8f acc[2][4];
#pragma unroll
    for (int i = 0; i < 2; ++i)
#pragma unroll
        for (int j = 0; j < 4; ++j) acc[i][j] = (v8f){0.f,0.f,0.f,0.f,0.f,0.f,0.f,0.f};

    int nk = K >> 5;
    stage_tiles(gA, K, gB, K, ldsA[0], ldsB[0], tid);
    tiles_ready_barrier();

    for (int kt = 0; kt < nk; ++kt) {
        int cur = kt & 1;
        if (kt + 1 < nk)
            stage_tiles(gA + (kt + 1) * 32, K, gB + (kt + 1) * 32, K,
                        ldsA[cur ^ 1], ldsB[cur ^ 1], tid);

        v16bf a0 = load_frag_lds(ldsA[cur], lm_base,      lane);
        v16bf a1 = load_frag_lds(ldsA[cur], lm_base + 16, lane);
        v16bf bf[4];
#pragma unroll
        for (int j = 0; j < 4; ++j)
            bf[j] = load_frag_lds(ldsB[cur], ln_base + 16 * j, lane);

#pragma unroll
        for (int j = 0; j < 4; ++j) {
            acc[0][j] = __builtin_amdgcn_wmma_f32_16x16x32_bf16(
                false, a0, false, bf[j], (short)0, acc[0][j], false, false);
            acc[1][j] = __builtin_amdgcn_wmma_f32_16x16x32_bf16(
                false, a1, false, bf[j], (short)0, acc[1][j], false, false);
        }

        if (kt + 1 < nk) tiles_ready_barrier();
    }

    // C layout: VGPR v, lanes 0-15 -> (M = v,   N = lane)
    //                   lanes16-31 -> (M = v+8, N = lane-16)
    int half = lane >> 4;
    int r    = lane & 15;
    int m_base = tm * 128 + lm_base;
    int n_base = tn * 128 + ln_base;
#pragma unroll
    for (int i = 0; i < 2; ++i) {
#pragma unroll
        for (int j = 0; j < 4; ++j) {
            int col = n_base + 16 * j + r;
            float bval = bias[col];
#pragma unroll
            for (int v = 0; v < 8; ++v) {
                int row = m_base + 16 * i + half * 8 + v;
                float val = acc[i][j][v] + bval;
                size_t idx = (size_t)row * Nn + col;
                if (MODE == 0) {
                    out_f32[idx] = val;
                } else if (MODE == 1) {
                    out_bf16[idx] = f2bf(val);
                } else { // MODE == 3
                    int b = row / S_;
                    float gate = modv[(size_t)b * THREED + 2 * D_ + col];
                    out_f32[idx] = x_res[idx] + gate * (val + skip[idx]);
                }
            }
        }
    }
}

// ---------------------------------------------------------------------------
// dbc = u_rev @ W_dbc.T  (u_rev = mzc time-reversed). 104 outputs per row.
// ---------------------------------------------------------------------------
__global__ void dbc_kernel(const float* __restrict__ u, const float* __restrict__ W_dbc,
                           float* __restrict__ delta_in, float* __restrict__ BC)
{
    int idx = blockIdx.x * blockDim.x + threadIdx.x;
    const int NOUT = R_ + 2 * N_; // 104
    if (idx >= M_ * NOUT) return;
    int rr = idx / NOUT, o = idx % NOUT;
    int b = rr / S_, sp = rr % S_;
    const float* urow = u + ((size_t)b * S_ + (S_ - 1 - sp)) * E_;
    const float* wrow = W_dbc + (size_t)o * E_;
    float acc = 0.f;
    for (int k = 0; k < E_; k += 4) {
        float4 a = *(const float4*)(urow + k);
        float4 w = *(const float4*)(wrow + k);
        acc += a.x * w.x + a.y * w.y + a.z * w.z + a.w * w.w;
    }
    if (o < R_) delta_in[(size_t)rr * R_ + o] = acc;
    else        BC[(size_t)rr * (2 * N_) + (o - R_)] = acc;
}

// ---------------------------------------------------------------------------
// delta = softplus(delta_in @ W_dt.T + b_dt)   (2048 x 2304, K=72)
// ---------------------------------------------------------------------------
__global__ void delta_kernel(const float* __restrict__ delta_in,
                             const float* __restrict__ W_dt, const float* __restrict__ b_dt,
                             float* __restrict__ delta)
{
    int idx = blockIdx.x * blockDim.x + threadIdx.x;
    if (idx >= M_ * E_) return;
    int rr = idx / E_, e = idx % E_;
    const float* din = delta_in + (size_t)rr * R_;
    const float* wrow = W_dt + (size_t)e * R_;
    float acc = b_dt[e];
#pragma unroll 4
    for (int r = 0; r < R_; ++r) acc += din[r] * wrow[r];
    float sp = (acc > 20.f) ? acc : log1pf(expf(acc));
    delta[idx] = sp;
}

// ---------------------------------------------------------------------------
// Sequential selective scan over reversed time; fused +u*D, un-reversal,
// silu(mx)*mz, bf16 output for the final GEMM.
// ---------------------------------------------------------------------------
__global__ __launch_bounds__(256) void scan_kernel(
    const float* __restrict__ delta, const float* __restrict__ BC,
    const float* __restrict__ u, const float* __restrict__ A_log,
    const float* __restrict__ D_ssm, const float* __restrict__ mx,
    __bf16* __restrict__ ybf)
{
    const int echunks = E_ / 256; // 9
    int b  = blockIdx.x / echunks;
    int ec = blockIdx.x % echunks;
    int e  = ec * 256 + threadIdx.x;

    __shared__ float sB[N_];
    __shared__ float sC[N_];

    float negA[N_];
#pragma unroll
    for (int n = 0; n < N_; ++n) negA[n] = -expf(A_log[(size_t)e * N_ + n]);
    float h[N_];
#pragma unroll
    for (int n = 0; n < N_; ++n) h[n] = 0.f;
    float Dv = D_ssm[e];

    for (int sp = 0; sp < S_; ++sp) {
        int rr   = b * S_ + sp;               // reversed-time row
        int orow = b * S_ + (S_ - 1 - sp);    // original-time row
        __syncthreads();
        if (threadIdx.x < 32) {
            float v = BC[(size_t)rr * (2 * N_) + threadIdx.x];
            if (threadIdx.x < N_) sB[threadIdx.x] = v;
            else                  sC[threadIdx.x - N_] = v;
        }
        __syncthreads();

        float d  = delta[(size_t)rr * E_ + e];
        float uu = u[(size_t)orow * E_ + e];
        float du = d * uu;
        float y = 0.f;
#pragma unroll
        for (int n = 0; n < N_; ++n) {
            h[n] = expf(d * negA[n]) * h[n] + du * sB[n];
            y += h[n] * sC[n];
        }
        float ys  = y + uu * Dv;
        float mxv = mx[(size_t)orow * E_ + e];
        ybf[(size_t)orow * E_ + e] = f2bf(siluf(mxv) * ys);
    }
}

// ---------------------------------------------------------------------------
// Launch
// ---------------------------------------------------------------------------
extern "C" void kernel_launch(void* const* d_in, const int* in_sizes, int n_in,
                              void* d_out, int out_size, void* d_ws, size_t ws_size,
                              hipStream_t stream)
{
    const float* x      = (const float*)d_in[0];
    const float* c      = (const float*)d_in[1];
    // d_in[2] = w (unused by reference)
    const float* W_ada  = (const float*)d_in[3];
    const float* b_ada  = (const float*)d_in[4];
    const float* g1     = (const float*)d_in[5];
    const float* be1    = (const float*)d_in[6];
    const float* g2     = (const float*)d_in[7];
    const float* be2    = (const float*)d_in[8];
    const float* W_x    = (const float*)d_in[9];
    const float* b_x    = (const float*)d_in[10];
    const float* W_z    = (const float*)d_in[11];
    const float* b_z    = (const float*)d_in[12];
    const float* W_conv = (const float*)d_in[13];
    const float* b_conv = (const float*)d_in[14];
    const float* W_dbc  = (const float*)d_in[15];
    const float* W_dt   = (const float*)d_in[16];
    const float* b_dt   = (const float*)d_in[17];
    const float* A_log  = (const float*)d_in[18];
    const float* D_ssm  = (const float*)d_in[19];
    const float* W_f    = (const float*)d_in[20];
    const float* b_f    = (const float*)d_in[21];
    float* out = (float*)d_out;

    // ---- carve workspace ----
    char* ws = (char*)d_ws;
    auto alloc = [&](size_t bytes) -> char* {
        char* p = ws;
        ws += (bytes + 255) & ~(size_t)255;
        return p;
    };
    float*  modv     = (float*)alloc((size_t)B_ * THREED * 4);
    __bf16* Wxb      = (__bf16*)alloc((size_t)E_ * D_ * 2);
    __bf16* Wzb      = (__bf16*)alloc((size_t)E_ * D_ * 2);
    __bf16* Wcb      = (__bf16*)alloc((size_t)E_ * E_ * 2);
    __bf16* Wfb      = (__bf16*)alloc((size_t)D_ * E_ * 2);
    float*  skip     = (float*)alloc((size_t)M_ * D_ * 4);
    __bf16* xn2b     = (__bf16*)alloc((size_t)M_ * D_ * 2);
    float*  mx       = (float*)alloc((size_t)M_ * E_ * 4);
    __bf16* mzb      = (__bf16*)alloc((size_t)M_ * E_ * 2);
    float*  mzc      = (float*)alloc((size_t)M_ * E_ * 4);
    float*  delta_in = (float*)alloc((size_t)M_ * R_ * 4);
    float*  BC       = (float*)alloc((size_t)M_ * 2 * N_ * 4);
    float*  delta    = (float*)alloc((size_t)M_ * E_ * 4);
    __bf16* ybf      = (__bf16*)alloc((size_t)M_ * E_ * 2);

    // 1) weight conversion to bf16
    cvt_kernel<<<1024, 256, 0, stream>>>(W_x,    Wxb, E_ * D_);
    cvt_kernel<<<1024, 256, 0, stream>>>(W_z,    Wzb, E_ * D_);
    cvt_kernel<<<2048, 256, 0, stream>>>(W_conv, Wcb, E_ * E_);
    cvt_kernel<<<1024, 256, 0, stream>>>(W_f,    Wfb, D_ * E_);

    // 2) adaLN modulation vector
    mod_kernel<<<(B_ * THREED + 255) / 256, 256, 0, stream>>>(c, W_ada, b_ada, modv);

    // 3) LN1 + modulate + LN2 (fused)
    ln_mod_kernel<<<M_, 256, 0, stream>>>(x, modv, g1, be1, g2, be2, skip, xn2b);

    // 4) mx = xn2 @ W_x.T + b_x  (f32 out)
    gemm_bf16_kernel<0><<<(M_ / 128) * (E_ / 128), 256, 0, stream>>>(
        xn2b, Wxb, M_, E_, D_, b_x, mx, nullptr, nullptr, nullptr, nullptr);
    // 5) mz = xn2 @ W_z.T + b_z  (bf16 out, feeds conv GEMM)
    gemm_bf16_kernel<1><<<(M_ / 128) * (E_ / 128), 256, 0, stream>>>(
        xn2b, Wzb, M_, E_, D_, b_z, nullptr, mzb, nullptr, nullptr, nullptr);
    // 6) mzc = mz @ W_conv.T + b_conv  (f32 out, = SSM input u)
    gemm_bf16_kernel<0><<<(M_ / 128) * (E_ / 128), 256, 0, stream>>>(
        mzb, Wcb, M_, E_, E_, b_conv, mzc, nullptr, nullptr, nullptr, nullptr);

    // 7) dbc projection (time-reversed) -> delta_in, B, C
    dbc_kernel<<<(M_ * (R_ + 2 * N_) + 255) / 256, 256, 0, stream>>>(
        mzc, W_dbc, delta_in, BC);
    // 8) delta = softplus(delta_in @ W_dt.T + b_dt)
    delta_kernel<<<(M_ * E_ + 255) / 256, 256, 0, stream>>>(delta_in, W_dt, b_dt, delta);

    // 9) selective scan + fused epilogue -> ybf (bf16)
    scan_kernel<<<B_ * (E_ / 256), 256, 0, stream>>>(
        delta, BC, mzc, A_log, D_ssm, mx, ybf);

    // 10) out = x + gate * (y @ W_f.T + b_f + skip)
    gemm_bf16_kernel<3><<<(M_ / 128) * (D_ / 128), 256, 0, stream>>>(
        ybf, Wfb, M_, D_, E_, b_f, out, nullptr, x, modv, skip);
}
